// MultiHeadAttention_38809324486924
// MI455X (gfx1250) — matrix operand order repaired and checked by reading, not compiled, not run on hardware
//
#include <hip/hip_runtime.h>
#include <math.h>

// ---------------------------------------------------------------------------
// MI455X / gfx1250 multi-head attention
//
//   Projections (~300 GFLOP, dominant): split-fp32 bf16x3 via
//     V_WMMA_F32_16X16X32_BF16  (X = Xh + Xl;  X@W^T ~ XhWh + XhWl + XlWh)
//     with emb/W/ctx PRE-SPLIT into pair-packed bf16 hi/lo planes so the
//     GEMM hot loop is pure b128 copy + WMMA (no per-slab conversion VALU).
//   Attention (QK^T, softmax, PV): native fp32 via V_WMMA_F32_16X16X4_F32
// ---------------------------------------------------------------------------

typedef __attribute__((ext_vector_type(2)))  float  v2f;
typedef __attribute__((ext_vector_type(8)))  float  v8f;
typedef __attribute__((ext_vector_type(16))) __bf16 v16bf;

#define D_MODEL 1024
#define DPAIR   512       // D_MODEL / 2 (pair-packed bf16 dwords per row)
#define NHEAD   16
#define DHEAD   64
#define NBATCH  32
#define QLEN    128
#define KLEN    2048

#define CHUNK        512  // softmax chunk (cols of logits held in LDS)
#define KT_PER_CHUNK 32
#define KT_PER_WAVE  8
#define NCHUNK       4

// ---- fp32 WMMA: D = A(16x4) * B(4x16) + C ---------------------------------
__device__ inline v8f wmma_f32(v2f a, v2f b, v8f c) {
  return __builtin_amdgcn_wmma_f32_16x16x4_f32(
      false, a, false, b, (short)0, c, false, false);
}
// ---- bf16 WMMA: D = A(16x32) * B(32x16) + C -------------------------------
__device__ inline v8f wmma_bf16(v16bf a, v16bf b, v8f c) {
  return __builtin_amdgcn_wmma_f32_16x16x32_bf16(
      false, a, false, b, (short)0, c, false, false);
}

// round-to-nearest-even fp32 -> bf16 bits
__device__ inline uint32_t bf16_bits(float x) {
  uint32_t u = __float_as_uint(x);
  return ((u + 0x7FFFu + ((u >> 16) & 1u)) >> 16) & 0xFFFFu;
}
__device__ inline float bf16_val(uint32_t b) { return __uint_as_float(b << 16); }
__device__ inline void split_pack(float x0, float x1, uint32_t& hp, uint32_t& lp) {
  uint32_t h0 = bf16_bits(x0), h1 = bf16_bits(x1);
  hp = (h1 << 16) | h0;
  uint32_t l0 = bf16_bits(x0 - bf16_val(h0));
  uint32_t l1 = bf16_bits(x1 - bf16_val(h1));
  lp = (l1 << 16) | l0;
}

union frag16 {
  uint32_t u[8];
  uint4    q[2];
  v16bf    v;
};

// ---------------------------------------------------------------------------
// Pre-split: f32[4i..4i+3] -> 2 pair-packed dwords in hi plane + 2 in lo.
// ---------------------------------------------------------------------------
__global__ __launch_bounds__(256)
void split_kernel(const float* __restrict__ src, uint32_t* __restrict__ hp,
                  uint32_t* __restrict__ lp, int nquads) {
  const int i = blockIdx.x * 256 + threadIdx.x;
  if (i >= nquads) return;
  const float4 v = *(const float4*)(src + (size_t)i * 4);
  uint32_t h0, l0, h1, l1;
  split_pack(v.x, v.y, h0, l0);
  split_pack(v.z, v.w, h1, l1);
  hp[(size_t)i * 2 + 0] = h0;
  hp[(size_t)i * 2 + 1] = h1;
  lp[(size_t)i * 2 + 0] = l0;
  lp[(size_t)i * 2 + 1] = l1;
}

// ---------------------------------------------------------------------------
// Y[N,1024] (fp32) = A @ W^T with A, W given as pre-split hi/lo planes
// (pair-packed bf16, row pitch DPAIR dwords). Optional row gather via idx;
// optional per-row output scale (mask * scale hoisted past the GEMM).
// block = 128 threads (4 waves), 64x64 tile, K slab = 64 (2 WMMA substeps).
// ---------------------------------------------------------------------------
__global__ __launch_bounds__(128)
void gemm_split_kernel(const uint32_t* __restrict__ Ah,
                       const uint32_t* __restrict__ Al,
                       const int*      __restrict__ idx,
                       const float*    __restrict__ rowscale,
                       const uint32_t* __restrict__ Wh,
                       const uint32_t* __restrict__ Wl,
                       float*          __restrict__ Y,
                       float scale) {
  // [row or col][pair 0..31], pitch 36 dwords (16B-aligned, conflict-free)
  __shared__ __align__(16) uint32_t sxh[64][36];
  __shared__ __align__(16) uint32_t sxl[64][36];
  __shared__ __align__(16) uint32_t swh[64][36];
  __shared__ __align__(16) uint32_t swl[64][36];

  const int tid  = threadIdx.x;
  const int lane = tid & 31;
  const int wave = tid >> 5;
  const int hi   = (lane >= 16) ? 1 : 0;
  const int mrow = lane & 15;

  const int nb = blockIdx.x * 64;   // output col base
  const int mb = blockIdx.y * 64;   // output row base

  v8f acc[4] = {};

  // staging roles: 2 threads per row/col, 16 pair-dwords each
  const int sr = tid >> 1;          // 0..63
  const int sp = (tid & 1) * 16;    // pair offset within 32-pair slab
  const int arow = idx ? idx[mb + sr] : (mb + sr);
  const uint32_t* axh = Ah + (size_t)arow * DPAIR + sp;
  const uint32_t* axl = Al + (size_t)arow * DPAIR + sp;
  const uint32_t* awh = Wh + (size_t)(nb + sr) * DPAIR + sp;
  const uint32_t* awl = Wl + (size_t)(nb + sr) * DPAIR + sp;

  for (int kp = 0; kp < DPAIR; kp += 32) {   // 16 slabs of K=64
    // ---- stage: pure b128 copies, no conversion --------------------------
    {
      uint4 a0 = *(const uint4*)(axh + kp + 0);
      uint4 a1 = *(const uint4*)(axh + kp + 4);
      uint4 a2 = *(const uint4*)(axh + kp + 8);
      uint4 a3 = *(const uint4*)(axh + kp + 12);
      *(uint4*)&sxh[sr][sp + 0]  = a0;
      *(uint4*)&sxh[sr][sp + 4]  = a1;
      *(uint4*)&sxh[sr][sp + 8]  = a2;
      *(uint4*)&sxh[sr][sp + 12] = a3;
      uint4 b0 = *(const uint4*)(axl + kp + 0);
      uint4 b1 = *(const uint4*)(axl + kp + 4);
      uint4 b2 = *(const uint4*)(axl + kp + 8);
      uint4 b3 = *(const uint4*)(axl + kp + 12);
      *(uint4*)&sxl[sr][sp + 0]  = b0;
      *(uint4*)&sxl[sr][sp + 4]  = b1;
      *(uint4*)&sxl[sr][sp + 8]  = b2;
      *(uint4*)&sxl[sr][sp + 12] = b3;
      uint4 c0 = *(const uint4*)(awh + kp + 0);
      uint4 c1 = *(const uint4*)(awh + kp + 4);
      uint4 c2 = *(const uint4*)(awh + kp + 8);
      uint4 c3 = *(const uint4*)(awh + kp + 12);
      *(uint4*)&swh[sr][sp + 0]  = c0;
      *(uint4*)&swh[sr][sp + 4]  = c1;
      *(uint4*)&swh[sr][sp + 8]  = c2;
      *(uint4*)&swh[sr][sp + 12] = c3;
      uint4 d0 = *(const uint4*)(awl + kp + 0);
      uint4 d1 = *(const uint4*)(awl + kp + 4);
      uint4 d2 = *(const uint4*)(awl + kp + 8);
      uint4 d3 = *(const uint4*)(awl + kp + 12);
      *(uint4*)&swl[sr][sp + 0]  = d0;
      *(uint4*)&swl[sr][sp + 4]  = d1;
      *(uint4*)&swl[sr][sp + 8]  = d2;
      *(uint4*)&swl[sr][sp + 12] = d3;
      if (kp + 32 < DPAIR) {
        __builtin_prefetch(axh + kp + 32, 0, 0);
        __builtin_prefetch(awh + kp + 32, 0, 0);
      }
    }
    __syncthreads();

#pragma unroll
    for (int sub = 0; sub < 2; ++sub) {      // two K=32 substeps per slab
      const int base = sub * 16;
      // A frags: lo lanes pairs {0..3, 8..11}, hi lanes {4..7, 12..15}
      frag16 Ahf, Alf;
      {
        const int ar = wave * 16 + mrow;
        const int ab = base + (hi ? 4 : 0);
        Ahf.q[0] = *(const uint4*)&sxh[ar][ab];
        Ahf.q[1] = *(const uint4*)&sxh[ar][ab + 8];
        Alf.q[0] = *(const uint4*)&sxl[ar][ab];
        Alf.q[1] = *(const uint4*)&sxl[ar][ab + 8];
      }
#pragma unroll
      for (int ct = 0; ct < 4; ++ct) {
        const int n  = ct * 16 + mrow;
        const int bb = base + (hi ? 4 : 0);
        frag16 Bhf, Blf;
        Bhf.q[0] = *(const uint4*)&swh[n][bb];
        Bhf.q[1] = *(const uint4*)&swh[n][bb + 8];
        Blf.q[0] = *(const uint4*)&swl[n][bb];
        Blf.q[1] = *(const uint4*)&swl[n][bb + 8];
        acc[ct] = wmma_bf16(Ahf.v, Bhf.v, acc[ct]);
        acc[ct] = wmma_bf16(Ahf.v, Blf.v, acc[ct]);
        acc[ct] = wmma_bf16(Alf.v, Bhf.v, acc[ct]);
      }
    }
    __syncthreads();
  }

  // ---- write 16x64 strip per wave; apply hoisted row scale (mask) ----------
#pragma unroll
  for (int r = 0; r < 8; ++r) {
    const int row = mb + wave * 16 + r + hi * 8;
    const float rs = scale * (rowscale ? rowscale[row] : 1.0f);
#pragma unroll
    for (int ct = 0; ct < 4; ++ct) {
      const int col = nb + ct * 16 + mrow;
      Y[(size_t)row * D_MODEL + col] = acc[ct][r] * rs;
    }
  }
}

// ---------------------------------------------------------------------------
// Flash attention, one block per (b, h, 16-row q tile). 128 threads, 4 waves.
// fp32 WMMA for QK^T / PV, chunked online softmax. Context is written
// pre-split (bf16 hi/lo pair-packed) for the output projection.
// ---------------------------------------------------------------------------
__global__ __launch_bounds__(128)
void attn_kernel(const float* __restrict__ Qe,      // [B,QL,D] (pre-scaled)
                 const float* __restrict__ Doc,     // [B,KL,D]
                 uint32_t*    __restrict__ CtxH,    // [B*QL, DPAIR]
                 uint32_t*    __restrict__ CtxL) {
  __shared__ float ldsL[16][CHUNK + 4];
  __shared__ float ldsK[4][16][DHEAD + 4];
  __shared__ float stat_m[16], stat_l[16], stat_f[16];
  __shared__ float red[16][8];

  const int tid  = threadIdx.x;
  const int lane = tid & 31;
  const int wave = tid >> 5;
  const int hi   = (lane >= 16) ? 1 : 0;
  const int mrow = lane & 15;

  const int qt = blockIdx.x;
  const int h  = blockIdx.y;
  const int b  = blockIdx.z;

  const float* qbase = Qe + ((size_t)b * QLEN + qt * 16) * D_MODEL + h * DHEAD;
  const float* dbase = Doc + (size_t)b * KLEN * D_MODEL + h * DHEAD;

  v2f qa[16];
#pragma unroll
  for (int s = 0; s < 16; ++s) {
    const int k0 = s * 4 + hi * 2;
    qa[s].x = qbase[(size_t)mrow * D_MODEL + k0];
    qa[s].y = qbase[(size_t)mrow * D_MODEL + k0 + 1];
  }

  if (tid < 16) { stat_m[tid] = -INFINITY; stat_l[tid] = 0.0f; }
  v8f octx[4] = {};
  __syncthreads();

  for (int ch = 0; ch < NCHUNK; ++ch) {
    // ---- pass 1: logits chunk -> ldsL --------------------------------------
    for (int t = 0; t < KT_PER_WAVE; ++t) {
      const int ltile = wave * KT_PER_WAVE + t;
      const int kt    = ch * KT_PER_CHUNK + ltile;
      const float* dt = dbase + (size_t)kt * 16 * D_MODEL;
      for (int e = lane; e < 256; e += 32) {
        const int r = e >> 4, c4 = (e & 15) * 4;
        float4 v = *(const float4*)(dt + (size_t)r * D_MODEL + c4);
        ldsK[wave][r][c4 + 0] = v.x;
        ldsK[wave][r][c4 + 1] = v.y;
        ldsK[wave][r][c4 + 2] = v.z;
        ldsK[wave][r][c4 + 3] = v.w;
      }
      __syncthreads();
      v8f cl = {};
#pragma unroll
      for (int s = 0; s < 16; ++s) {
        const int k0 = s * 4 + hi * 2;
        v2f bf;
        bf.x = ldsK[wave][mrow][k0];
        bf.y = ldsK[wave][mrow][k0 + 1];
        cl = wmma_f32(qa[s], bf, cl);
      }
#pragma unroll
      for (int r = 0; r < 8; ++r)
        ldsL[r + hi * 8][ltile * 16 + mrow] = cl[r];
      __syncthreads();
    }

    // ---- online softmax ----------------------------------------------------
    const int row = tid & 15, seg = tid >> 4;
    float pm = -INFINITY;
    for (int c = seg * 64; c < seg * 64 + 64; ++c) pm = fmaxf(pm, ldsL[row][c]);
    red[row][seg] = pm;
    __syncthreads();
    if (tid < 16) {
      float m0 = stat_m[tid], nm = m0;
      for (int s = 0; s < 8; ++s) nm = fmaxf(nm, red[tid][s]);
      stat_f[tid] = __expf(m0 - nm);
      stat_m[tid] = nm;
    }
    __syncthreads();
    {
      const float nm = stat_m[row];
      float ps = 0.0f;
      for (int c = seg * 64; c < seg * 64 + 64; ++c) {
        const float e = __expf(ldsL[row][c] - nm);
        ldsL[row][c] = e;
        ps += e;
      }
      red[row][seg] = ps;
    }
    __syncthreads();
    if (tid < 16) {
      float s = 0.0f;
      for (int i = 0; i < 8; ++i) s += red[tid][i];
      stat_l[tid] = stat_l[tid] * stat_f[tid] + s;
    }
    __syncthreads();

#pragma unroll
    for (int ct = 0; ct < 4; ++ct)
#pragma unroll
      for (int r = 0; r < 8; ++r)
        octx[ct][r] *= stat_f[r + hi * 8];

    // ---- pass 2: octx += P_hat @ V ----------------------------------------
    for (int t = 0; t < KT_PER_WAVE; ++t) {
      const int ltile = wave * KT_PER_WAVE + t;
      const int kt    = ch * KT_PER_CHUNK + ltile;
      const float* dt = dbase + (size_t)kt * 16 * D_MODEL;
      for (int e = lane; e < 256; e += 32) {
        const int r = e >> 4, c4 = (e & 15) * 4;
        float4 v = *(const float4*)(dt + (size_t)r * D_MODEL + c4);
        ldsK[wave][r][c4 + 0] = v.x;
        ldsK[wave][r][c4 + 1] = v.y;
        ldsK[wave][r][c4 + 2] = v.z;
        ldsK[wave][r][c4 + 3] = v.w;
      }
      __syncthreads();
#pragma unroll
      for (int s = 0; s < 4; ++s) {
        const int k0 = s * 4 + hi * 2;
        v2f a;
        a.x = ldsL[mrow][ltile * 16 + k0];
        a.y = ldsL[mrow][ltile * 16 + k0 + 1];
#pragma unroll
        for (int ct = 0; ct < 4; ++ct) {
          v2f bb;
          bb.x = ldsK[wave][k0][ct * 16 + mrow];
          bb.y = ldsK[wave][k0 + 1][ct * 16 + mrow];
          octx[ct] = wmma_f32(a, bb, octx[ct]);
        }
      }
      __syncthreads();
    }
  }

  // ---- cross-wave reduction + split-packed context write -------------------
  float* ctxred = &ldsL[0][0];               // 4*16*64 floats, aliases ldsL
  __syncthreads();
#pragma unroll
  for (int ct = 0; ct < 4; ++ct)
#pragma unroll
    for (int r = 0; r < 8; ++r)
      ctxred[((wave * 16) + r + hi * 8) * 64 + ct * 16 + mrow] = octx[ct][r];
  __syncthreads();
  for (int e = tid; e < 512; e += 128) {     // 16 rows x 32 pairs
    const int row = e >> 5, pr = e & 31;
    const int c0 = pr * 2;
    const float inv = 1.0f / stat_l[row];
    const float s0 = (ctxred[(0 * 16 + row) * 64 + c0] +
                      ctxred[(1 * 16 + row) * 64 + c0] +
                      ctxred[(2 * 16 + row) * 64 + c0] +
                      ctxred[(3 * 16 + row) * 64 + c0]) * inv;
    const float s1 = (ctxred[(0 * 16 + row) * 64 + c0 + 1] +
                      ctxred[(1 * 16 + row) * 64 + c0 + 1] +
                      ctxred[(2 * 16 + row) * 64 + c0 + 1] +
                      ctxred[(3 * 16 + row) * 64 + c0 + 1]) * inv;
    uint32_t hp, lp;
    split_pack(s0, s1, hp, lp);
    const size_t po =
        ((size_t)b * QLEN + qt * 16 + row) * DPAIR + h * (DHEAD / 2) + pr;
    CtxH[po] = hp;
    CtxL[po] = lp;
  }
}

// ---------------------------------------------------------------------------
extern "C" void kernel_launch(void* const* d_in, const int* in_sizes, int n_in,
                              void* d_out, int out_size, void* d_ws,
                              size_t ws_size, hipStream_t stream) {
  const int*   q       = (const int*)d_in[0];
  const int*   d_pos   = (const int*)d_in[1];
  const int*   d_neg   = (const int*)d_in[2];
  const float* q_mask  = (const float*)d_in[3];
  const float* dp_mask = (const float*)d_in[4];
  const float* dn_mask = (const float*)d_in[5];
  const float* emb     = (const float*)d_in[6];
  const float* W_trans = (const float*)d_in[7];
  const float* W_out   = (const float*)d_in[8];
  float*       out     = (float*)d_out;

  // ---- workspace carve (dword units) ---------------------------------------
  const size_t EMB_P = (size_t)32000 * DPAIR;   // 16,384,000
  const size_t W_P   = (size_t)1024 * DPAIR;    //    524,288
  const size_t CTX_P = (size_t)4096 * DPAIR;    //  2,097,152
  uint32_t* emb_h = (uint32_t*)d_ws;
  uint32_t* emb_l = emb_h + EMB_P;
  uint32_t* wt_h  = emb_l + EMB_P;
  uint32_t* wt_l  = wt_h + W_P;
  uint32_t* wo_h  = wt_l + W_P;
  uint32_t* wo_l  = wo_h + W_P;
  uint32_t* ctx_h = wo_l + W_P;
  uint32_t* ctx_l = ctx_h + CTX_P;
  float*    qe    = (float*)(ctx_l + CTX_P);    // 4096 x 1024 fp32
  float*    doc   = qe + (size_t)4096 * 1024;   // 65536 x 1024 fp32

  const dim3 blk128(128);
  const float SCALE = 0.125f;   // 64^-0.5

  // ---- pre-split emb table and weight matrices (once per launch) -----------
  {
    const int emb_quads = 32000 * D_MODEL / 4;  // 8,192,000
    const int w_quads   = 1024 * D_MODEL / 4;   //   262,144
    split_kernel<<<(emb_quads + 255) / 256, 256, 0, stream>>>(emb, emb_h,
                                                              emb_l, emb_quads);
    split_kernel<<<(w_quads + 255) / 256, 256, 0, stream>>>(W_trans, wt_h,
                                                            wt_l, w_quads);
    split_kernel<<<(w_quads + 255) / 256, 256, 0, stream>>>(W_out, wo_h, wo_l,
                                                            w_quads);
  }

  // queries: gather + W_trans^T projection (mask & 1/sqrt(dh) hoisted to out)
  gemm_split_kernel<<<dim3(16, 4096 / 64), blk128, 0, stream>>>(
      emb_h, emb_l, q, q_mask, wt_h, wt_l, qe, SCALE);

  for (int d = 0; d < 2; ++d) {
    const int*   didx  = d ? d_neg : d_pos;
    const float* dmask = d ? dn_mask : dp_mask;

    gemm_split_kernel<<<dim3(16, 65536 / 64), blk128, 0, stream>>>(
        emb_h, emb_l, didx, dmask, wt_h, wt_l, doc, 1.0f);

    attn_kernel<<<dim3(QLEN / 16, NHEAD, NBATCH), blk128, 0, stream>>>(
        qe, doc, ctx_h, ctx_l);

    gemm_split_kernel<<<dim3(16, 4096 / 64), blk128, 0, stream>>>(
        ctx_h, ctx_l, nullptr, nullptr, wo_h, wo_l,
        out + (size_t)d * 4096 * 1024, 1.0f);
  }
}
